// GatedGCN_66683662238132
// MI455X (gfx1250) — matrix operand order repaired and checked by reading, not compiled, not run on hardware
//
#include <hip/hip_runtime.h>
#include <hip/hip_bf16.h>
#include <math.h>

#define NNODES 100000
#define NEDGES 1250000
#define IN_DIM 128
#define HID 64
#define MOTIF 16

typedef __attribute__((ext_vector_type(16))) __bf16        v16bf;
typedef __attribute__((ext_vector_type(8)))  float         v8f;
typedef __attribute__((ext_vector_type(4)))  float         v4f;
typedef __attribute__((ext_vector_type(4)))  unsigned int  v4u;
typedef __attribute__((ext_vector_type(2)))  unsigned int  v2u;

union BFrag { v16bf v; v4u q[2]; unsigned short u[16]; };

__device__ __forceinline__ unsigned short f2bf(float f) {
  unsigned int x = __float_as_uint(f);
  x += 0x7FFFu + ((x >> 16) & 1u);               // round-to-nearest-even
  return (unsigned short)(x >> 16);
}
__device__ __forceinline__ unsigned int pack2(float a, float b) {
  return (unsigned int)f2bf(a) | ((unsigned int)f2bf(b) << 16);
}

// ------------------------------------------------------------- weight swizzle
// Store W[K][64] as WMMA B fragments: frag f=(k0/32)*4+nt, lane l, halves h=0..15
// contiguous at dst[((f*32+l)*16)+h], h -> K = k0 + h + (l>=16?16:0), col = nt*16+(l&15)
__global__ void k_w_swizzle(const float* __restrict__ W,
                            unsigned short* __restrict__ dst, int K) {
  int i = blockIdx.x * blockDim.x + threadIdx.x;
  if (i >= K * 64) return;
  int h    = i & 15;
  int lane = (i >> 4) & 31;
  int f    = i >> 9;
  int s    = f >> 2, nt = f & 3;
  int col  = nt * 16 + (lane & 15);
  int k    = s * 32 + h + ((lane >> 4) ? 16 : 0);
  dst[i] = f2bf(W[k * 64 + col]);
}

// ---------------------------------------------------------------- edge gate
// One wave = 16 edges. feats[16][64] bf16 in LDS; B frags from swizzled global.
__global__ void k_edge_gate(const float* __restrict__ motif,
                            const long long* __restrict__ ei,
                            const unsigned short* __restrict__ gW1sw,
                            const float* __restrict__ gb1,
                            const float* __restrict__ gW2,
                            const float* __restrict__ gb2,
                            float* __restrict__ gate,
                            int nTiles) {
  __shared__ unsigned short sF[8][16 * 64];
  const int tid  = threadIdx.x;
  const int wave = tid >> 5;
  const int lane = tid & 31;
  const int l16  = lane & 15;
  const int hi   = lane >> 4;

  float w2[4], b1v[4];
#pragma unroll
  for (int nt = 0; nt < 4; ++nt) {
    w2[nt]  = gW2[nt * 16 + l16];
    b1v[nt] = gb1[nt * 16 + l16];
  }
  const float bias2 = gb2[0];

  int tile = blockIdx.x * 8 + wave;
  if (tile >= nTiles) return;
  long long ebase = (long long)tile * 16;

  // Stage feats: lane handles edge (lane&15); hi=0 -> feats[0..31], hi=1 -> [32..63]
  {
    long long e = ebase + l16;
    long long s = ei[e];
    long long d = ei[(long long)NEDGES + e];
    const v4f* mu4 = (const v4f*)(motif + s * MOTIF);
    const v4f* mv4 = (const v4f*)(motif + d * MOTIF);
    unsigned int* row32 = (unsigned int*)&sF[wave][l16 * 64];
#pragma unroll
    for (int c = 0; c < 4; ++c) {
      v4f a = mu4[c], b = mv4[c];
      if (hi == 0) {
        row32[c * 2]      = pack2(a.x, a.y);
        row32[c * 2 + 1]  = pack2(a.z, a.w);
        row32[8 + c * 2]     = pack2(b.x, b.y);
        row32[8 + c * 2 + 1] = pack2(b.z, b.w);
      } else {
        row32[16 + c * 2]     = pack2(fabsf(a.x - b.x), fabsf(a.y - b.y));
        row32[16 + c * 2 + 1] = pack2(fabsf(a.z - b.z), fabsf(a.w - b.w));
        row32[24 + c * 2]     = pack2(a.x * b.x, a.y * b.y);
        row32[24 + c * 2 + 1] = pack2(a.z * b.z, a.w * b.w);
      }
    }
  }
  // wave-synchronous LDS (single wave wrote + reads its own tile)

  v8f acc[4] = {};
#pragma unroll
  for (int k0 = 0; k0 < 64; k0 += 32) {
    BFrag a;
    const v4u* ap = (const v4u*)&sF[wave][l16 * 64 + k0 + hi * 16];
    a.q[0] = ap[0];
    a.q[1] = ap[1];                                // +8 halves
#pragma unroll
    for (int nt = 0; nt < 4; ++nt) {
      BFrag b;
      const v4u* bp =
          (const v4u*)(gW1sw + ((size_t)(((k0 >> 5) * 4 + nt) * 32 + lane) << 4));
      b.q[0] = bp[0];
      b.q[1] = bp[1];
      acc[nt] = __builtin_amdgcn_wmma_f32_16x16x32_bf16(
          false, a.v, false, b.v, (short)0, acc[nt], false, false);
    }
  }

  // relu + second layer (64 -> 1): in-lane partials + 16-lane butterfly
  float p[8];
#pragma unroll
  for (int r = 0; r < 8; ++r) {
    float s = 0.f;
#pragma unroll
    for (int nt = 0; nt < 4; ++nt) {
      float g = acc[nt][r] + b1v[nt];
      g = g > 0.f ? g : 0.f;
      s += g * w2[nt];
    }
    p[r] = s;
  }
#pragma unroll
  for (int m = 1; m < 16; m <<= 1) {
#pragma unroll
    for (int r = 0; r < 8; ++r) p[r] += __shfl_xor(p[r], m, 32);
  }
  if (l16 == 0) {                                  // lanes 0 and 16 write 8 edges each
#pragma unroll
    for (int r = 0; r < 8; ++r) {
      float v = p[r] + bias2;
      float g = 1.f / (1.f + __expf(-v));
      g = fminf(fmaxf(g, 0.f), 1.f);
      gate[ebase + hi * 8 + r] = g;
    }
  }
}

// ---------------------------------------------------------------- degrees
__global__ void k_deg_init(float* deg, int n) {
  int i = blockIdx.x * blockDim.x + threadIdx.x;
  if (i < n) deg[i] = 1.0f;                        // self-loop weight
}
__global__ void k_deg_acc(const long long* __restrict__ ei,
                          const float* __restrict__ gate, float* deg) {
  long long i = (long long)blockIdx.x * blockDim.x + threadIdx.x;
  if (i < NEDGES) atomicAdd(&deg[ei[(long long)NEDGES + i]], gate[i]);
}
__global__ void k_deg_to_dis(float* deg, int n) {
  int i = blockIdx.x * blockDim.x + threadIdx.x;
  if (i < n) {
    float d = deg[i];
    deg[i] = d > 0.f ? rsqrtf(fmaxf(d, 1e-12f)) : 0.f;
  }
}

// ---------------------------------------------------------------- node GEMM (WMMA)
// out0 = X @ W [+bias]; if out1: out1 = out0 * dis[row]^2 (self-loop term)
template <int K>
__global__ void k_node_gemm(const float* __restrict__ X,
                            const unsigned short* __restrict__ Wsw,
                            const float* __restrict__ bias,
                            const float* __restrict__ dis,
                            float* __restrict__ out0,
                            float* __restrict__ out1,
                            int nTiles) {
  __shared__ unsigned short sX[8][16 * K];
  const int wave = threadIdx.x >> 5;
  const int lane = threadIdx.x & 31;
  const int l16  = lane & 15;
  const int hi   = lane >> 4;

  int tile = blockIdx.x * 8 + wave;
  if (tile >= nTiles) return;
  size_t rowBase = (size_t)tile * 16;

  // stage 16 x K tile as bf16 (contiguous rows): float4 load, 8B LDS store
  for (int i = lane * 4; i < 16 * K; i += 128) {
    v4f f = *(const v4f*)(X + rowBase * K + i);
    v2u pk;
    pk.x = pack2(f.x, f.y);
    pk.y = pack2(f.z, f.w);
    *(v2u*)&sX[wave][i] = pk;
  }

  v8f acc[4] = {};
#pragma unroll
  for (int k0 = 0; k0 < K; k0 += 32) {
    BFrag a;
    const v4u* ap = (const v4u*)&sX[wave][l16 * K + k0 + hi * 16];
    a.q[0] = ap[0];
    a.q[1] = ap[1];                                // +8 halves
#pragma unroll
    for (int nt = 0; nt < 4; ++nt) {
      BFrag b;
      const v4u* bp =
          (const v4u*)(Wsw + ((size_t)(((k0 >> 5) * 4 + nt) * 32 + lane) << 4));
      b.q[0] = bp[0];
      b.q[1] = bp[1];
      acc[nt] = __builtin_amdgcn_wmma_f32_16x16x32_bf16(
          false, a.v, false, b.v, (short)0, acc[nt], false, false);
    }
  }

#pragma unroll
  for (int nt = 0; nt < 4; ++nt) {
#pragma unroll
    for (int r = 0; r < 8; ++r) {
      int M = r + (hi ? 8 : 0);
      size_t row = rowBase + M;
      int col = nt * 16 + l16;
      float v = acc[nt][r];
      if (bias) v += bias[col];
      size_t idx = row * 64 + col;
      out0[idx] = v;
      if (out1) { float ds = dis[row]; out1[idx] = v * ds * ds; }
    }
  }
}

// ---------------------------------------------------------------- edge scatter
__global__ void k_scatter(const long long* __restrict__ ei,
                          const float* __restrict__ gate,
                          const float* __restrict__ dis,
                          const float* __restrict__ xw,
                          float* __restrict__ h) {
  long long idx = (long long)blockIdx.x * blockDim.x + threadIdx.x;
  if (idx >= (long long)NEDGES * 8) return;
  long long e = idx >> 3;
  int j0 = ((int)idx & 7) * 8;
  long long s = ei[e];
  long long d = ei[(long long)NEDGES + e];
  float w = dis[s] * gate[e] * dis[d];
  const v4f* src = (const v4f*)(xw + s * 64 + j0);
  v4f a = src[0], b = src[1];
  float* dst = h + d * 64 + j0;
  atomicAdd(dst + 0, a.x * w); atomicAdd(dst + 1, a.y * w);
  atomicAdd(dst + 2, a.z * w); atomicAdd(dst + 3, a.w * w);
  atomicAdd(dst + 4, b.x * w); atomicAdd(dst + 5, b.y * w);
  atomicAdd(dst + 6, b.z * w); atomicAdd(dst + 7, b.w * w);
}

// ---------------------------------------------------------------- LN + ReLU + residual
__global__ void k_node_ln(const float* __restrict__ h,
                          const float* __restrict__ bias,
                          const float* __restrict__ g,
                          const float* __restrict__ b,
                          const float* __restrict__ resid,
                          const float* __restrict__ resid_b,
                          float* __restrict__ out, int n) {
  int wave = threadIdx.x >> 5, lane = threadIdx.x & 31;
  int node = blockIdx.x * (blockDim.x >> 5) + wave;
  if (node >= n) return;
  size_t base = (size_t)node * 64;
  float v0 = h[base + lane] + bias[lane];
  float v1 = h[base + lane + 32] + bias[lane + 32];
  float s = v0 + v1, sq = v0 * v0 + v1 * v1;
#pragma unroll
  for (int m = 1; m < 32; m <<= 1) {
    s  += __shfl_xor(s, m, 32);
    sq += __shfl_xor(sq, m, 32);
  }
  float mean = s * (1.f / 64.f);
  float var  = sq * (1.f / 64.f) - mean * mean;
  float inv  = rsqrtf(var + 1e-5f);
  float o0 = fmaxf((v0 - mean) * inv * g[lane]      + b[lane],      0.f);
  float o1 = fmaxf((v1 - mean) * inv * g[lane + 32] + b[lane + 32], 0.f);
  float r0 = resid[base + lane]      + o0;
  float r1 = resid[base + lane + 32] + o1;
  if (resid_b) { r0 += resid_b[lane]; r1 += resid_b[lane + 32]; }
  out[base + lane]      = r0;
  out[base + lane + 32] = r1;
}

// ---------------------------------------------------------------- launch
extern "C" void kernel_launch(void* const* d_in, const int* in_sizes, int n_in,
                              void* d_out, int out_size, void* d_ws, size_t ws_size,
                              hipStream_t stream) {
  (void)in_sizes; (void)n_in; (void)out_size; (void)ws_size;
  const float*     x     = (const float*)d_in[0];
  const float*     motif = (const float*)d_in[1];
  const long long* ei    = (const long long*)d_in[2];
  const float* gW1  = (const float*)d_in[3];
  const float* gb1  = (const float*)d_in[4];
  const float* gW2  = (const float*)d_in[5];
  const float* gb2  = (const float*)d_in[6];
  const float* W0   = (const float*)d_in[7];
  const float* b0   = (const float*)d_in[8];
  const float* W1   = (const float*)d_in[9];
  const float* b1   = (const float*)d_in[10];
  const float* ln0g = (const float*)d_in[11];
  const float* ln0b = (const float*)d_in[12];
  const float* ln1g = (const float*)d_in[13];
  const float* ln1b = (const float*)d_in[14];
  const float* res0W = (const float*)d_in[15];
  const float* res0b = (const float*)d_in[16];
  const float* headW = (const float*)d_in[17];
  const float* headb = (const float*)d_in[18];
  float* out = (float*)d_out;

  char* ws = (char*)d_ws;
  size_t off = 0;
  auto alloc = [&](size_t bytes) {
    char* p = ws + off;
    off = (off + bytes + 255) & ~(size_t)255;
    return p;
  };
  float* gate = (float*)alloc((size_t)NEDGES * 4);
  float* dis  = (float*)alloc((size_t)NNODES * 4);
  float* bufA = (float*)alloc((size_t)NNODES * 64 * 4);   // xw0 -> xw1 -> x2
  float* bufH = (float*)alloc((size_t)NNODES * 64 * 4);   // h   -> h2
  float* bufR = (float*)alloc((size_t)NNODES * 64 * 4);   // xr  -> x1
  unsigned short* gW1sw  = (unsigned short*)alloc(64 * 64 * 2);
  unsigned short* W0sw   = (unsigned short*)alloc(128 * 64 * 2);
  unsigned short* res0sw = (unsigned short*)alloc(128 * 64 * 2);
  unsigned short* W1sw   = (unsigned short*)alloc(64 * 64 * 2);
  unsigned short* headsw = (unsigned short*)alloc(64 * 64 * 2);

  // weights -> WMMA-B-fragment-swizzled bf16
  k_w_swizzle<<<(4096 + 255) / 256, 256, 0, stream>>>(gW1,   gW1sw,  64);
  k_w_swizzle<<<(8192 + 255) / 256, 256, 0, stream>>>(W0,    W0sw,   128);
  k_w_swizzle<<<(8192 + 255) / 256, 256, 0, stream>>>(res0W, res0sw, 128);
  k_w_swizzle<<<(4096 + 255) / 256, 256, 0, stream>>>(W1,    W1sw,   64);
  k_w_swizzle<<<(4096 + 255) / 256, 256, 0, stream>>>(headW, headsw, 64);

  // edge gate (WMMA)
  const int eTiles = NEDGES / 16;                       // 78125
  k_edge_gate<<<(eTiles + 7) / 8, 256, 0, stream>>>(motif, ei, gW1sw, gb1, gW2, gb2,
                                                    gate, eTiles);
  // symmetric normalization
  k_deg_init<<<(NNODES + 255) / 256, 256, 0, stream>>>(dis, NNODES);
  k_deg_acc<<<(NEDGES + 255) / 256, 256, 0, stream>>>(ei, gate, dis);
  k_deg_to_dis<<<(NNODES + 255) / 256, 256, 0, stream>>>(dis, NNODES);

  const int nTiles = NNODES / 16;                       // 6250
  const int gB = (nTiles + 7) / 8;
  const long long sThreads = (long long)NEDGES * 8;
  const int sB = (int)((sThreads + 255) / 256);

  // layer 0
  k_node_gemm<128><<<gB, 256, 0, stream>>>(x, W0sw, nullptr, dis, bufA, bufH, nTiles);
  k_node_gemm<128><<<gB, 256, 0, stream>>>(x, res0sw, nullptr, nullptr, bufR, nullptr, nTiles);
  k_scatter<<<sB, 256, 0, stream>>>(ei, gate, dis, bufA, bufH);
  k_node_ln<<<(NNODES + 7) / 8, 256, 0, stream>>>(bufH, b0, ln0g, ln0b, bufR, res0b,
                                                  bufR, NNODES);
  // layer 1
  k_node_gemm<64><<<gB, 256, 0, stream>>>(bufR, W1sw, nullptr, dis, bufA, bufH, nTiles);
  k_scatter<<<sB, 256, 0, stream>>>(ei, gate, dis, bufA, bufH);
  k_node_ln<<<(NNODES + 7) / 8, 256, 0, stream>>>(bufH, b1, ln1g, ln1b, bufR, nullptr,
                                                  bufA, NNODES);
  // head
  k_node_gemm<64><<<gB, 256, 0, stream>>>(bufA, headsw, headb, nullptr, out, nullptr, nTiles);
}
